// LinearAttention_5334349382238
// MI455X (gfx1250) — matrix operand order repaired and checked by reading, not compiled, not run on hardware
//
#include <hip/hip_runtime.h>
#include <math.h>

// LinearAttention for MI455X (gfx1250): B=4, N=8192, C=1024, H=16, D=64.
// Heavy contractions on v_wmma_f32_16x16x32_f16 (wave32 WMMA, f32 acc).
// Memory-bound -> f16 intermediates; pure-copy tiles use the CDNA5 async
// global->LDS path (ASYNCcnt) via inline asm; converting loads stay on VGPRs.

typedef _Float16 half_t;
typedef __attribute__((ext_vector_type(16))) _Float16 v16h;
typedef __attribute__((ext_vector_type(8)))  _Float16 v8h;
typedef __attribute__((ext_vector_type(8)))  float    v8f;

// ---- CDNA5 async global->LDS copy (CDNA5 ISA) --------
// Each lane copies 16 bytes from its global address to its LDS byte address.
// LDS aperture maps addr[31:0] -> wave-relative LDS offset (ISA §10.2).
__device__ __forceinline__ void async_copy16(void* lds_ptr, const void* gptr) {
  uint32_t lds_off = (uint32_t)(uintptr_t)lds_ptr;
  uint64_t ga = (uint64_t)(uintptr_t)gptr;
  asm volatile("global_load_async_to_lds_b128 %0, %1, off"
               :: "v"(lds_off), "v"(ga) : "memory");
}
__device__ __forceinline__ void wait_async() {
  asm volatile("s_wait_asynccnt 0x0" ::: "memory");
}

// ---- WMMA fragment loaders (CDNA5 16-bit layouts, CDNA5 ISA) ----
__device__ __forceinline__ v16h frag_a(const half_t* s, int ld, int mt, int ks, int lane) {
  const half_t* p = s + (mt + (lane & 15)) * ld + ks + ((lane & 16) ? 8 : 0);
  v16h a;
#pragma unroll
  for (int i = 0; i < 8; ++i) { a[i] = p[i]; a[i + 8] = p[i + 16]; }
  return a;
}
__device__ __forceinline__ v16h frag_b(const half_t* s, int ld, int nt, int ks, int lane) {
  const half_t* p = s + (nt + (lane & 15)) * ld + ks + ((lane & 16) ? 8 : 0);
  v16h b;
#pragma unroll
  for (int i = 0; i < 8; ++i) { b[i] = p[i]; b[i + 8] = p[i + 16]; }
  return b;
}
__device__ __forceinline__ v8f wmma16(v16h a, v16h b, v8f c) {
  return __builtin_amdgcn_wmma_f32_16x16x32_f16(false, a, false, b, (short)0, c, false, false);
}
__device__ __forceinline__ v8f vzero8() {
  return (v8f){0.f, 0.f, 0.f, 0.f, 0.f, 0.f, 0.f, 0.f};
}

// ---------------- Kernel 1: QKV GEMM + bias + elu+1, scatter to q/kT/v ------
__global__ __launch_bounds__(256) void la_qkv_kernel(
    const float* __restrict__ x, const float* __restrict__ w,
    const float* __restrict__ bias, half_t* __restrict__ q,
    half_t* __restrict__ kT, half_t* __restrict__ v) {
  __shared__ half_t As[128 * 72];
  __shared__ half_t Bs[128 * 72];
  const int tid = threadIdx.x, lane = tid & 31, wid = tid >> 5;
  const int wm = wid >> 1, wn = wid & 1;
  const int m0 = blockIdx.y * 128, n0 = blockIdx.x * 128;

  v8f acc[2][4];
#pragma unroll
  for (int mi = 0; mi < 2; ++mi)
#pragma unroll
    for (int ni = 0; ni < 4; ++ni) acc[mi][ni] = vzero8();

  for (int kt = 0; kt < 1024; kt += 64) {
    __syncthreads();
    if (kt + 64 < 1024) {   // pull next weight k-tile toward GL2
      __builtin_prefetch(w + (size_t)(kt + 64 + (tid >> 5)) * 3072 + n0 + (tid & 31) * 4, 0, 1);
    }
#pragma unroll
    for (int p = 0; p < 8; ++p) {          // A tile: 128x64 f32 -> f16 LDS
      int f = p * 256 + tid;
      int row = f >> 4, c4 = f & 15;
      float4 xv = *(const float4*)(x + (size_t)(m0 + row) * 1024 + kt + c4 * 4);
      half_t* d = &As[row * 72 + c4 * 4];
      d[0] = (half_t)xv.x; d[1] = (half_t)xv.y; d[2] = (half_t)xv.z; d[3] = (half_t)xv.w;
    }
#pragma unroll
    for (int p = 0; p < 8; ++p) {          // B tile: 64x128 f32 -> N-major f16 LDS
      int f = p * 256 + tid;
      int kr = f >> 5, c4 = f & 31;
      float4 wv = *(const float4*)(w + (size_t)(kt + kr) * 3072 + n0 + c4 * 4);
      Bs[(c4 * 4 + 0) * 72 + kr] = (half_t)wv.x;
      Bs[(c4 * 4 + 1) * 72 + kr] = (half_t)wv.y;
      Bs[(c4 * 4 + 2) * 72 + kr] = (half_t)wv.z;
      Bs[(c4 * 4 + 3) * 72 + kr] = (half_t)wv.w;
    }
    __syncthreads();
#pragma unroll
    for (int ks = 0; ks < 64; ks += 32) {
      v16h af[2], bf[4];
#pragma unroll
      for (int mi = 0; mi < 2; ++mi) af[mi] = frag_a(As, 72, wm * 32 + mi * 16, ks, lane);
#pragma unroll
      for (int ni = 0; ni < 4; ++ni) bf[ni] = frag_b(Bs, 72, wn * 64 + ni * 16, ks, lane);
#pragma unroll
      for (int mi = 0; mi < 2; ++mi)
#pragma unroll
        for (int ni = 0; ni < 4; ++ni) acc[mi][ni] = wmma16(af[mi], bf[ni], acc[mi][ni]);
    }
  }
  // Epilogue: bias, elu+1 on q/k, scatter: q[bh,n,d], kT[bh,d,n], v[bh,n,d]
#pragma unroll
  for (int mi = 0; mi < 2; ++mi)
#pragma unroll
    for (int ni = 0; ni < 4; ++ni)
#pragma unroll
      for (int r = 0; r < 8; ++r) {
        int row = m0 + wm * 32 + mi * 16 + r + ((lane & 16) ? 8 : 0);
        int col = n0 + wn * 64 + ni * 16 + (lane & 15);
        float val = acc[mi][ni][r] + bias[col];
        int which = col >> 10, hc = col & 1023;
        int h = hc >> 6, dd = hc & 63;
        int b = row >> 13, n = row & 8191;
        size_t bh = (size_t)b * 16 + h;
        if (which == 0) {
          float f = val > 0.f ? val + 1.f : __expf(val);   // elu(x)+1
          q[(bh * 8192 + n) * 64 + dd] = (half_t)f;
        } else if (which == 1) {
          float f = val > 0.f ? val + 1.f : __expf(val);
          kT[(bh * 64 + dd) * 8192 + n] = (half_t)f;       // transposed store
        } else {
          v[(bh * 8192 + n) * 64 + dd] = (half_t)val;
        }
      }
}

// ---------------- Kernel 2: ksum[bh,d] = sum_n k ----------------------------
__global__ __launch_bounds__(256) void la_ksum_kernel(
    const half_t* __restrict__ kT, float* __restrict__ ksum) {
  __shared__ float red[256];
  const int row = blockIdx.x;                  // bh*64 + d, 4096 rows
  const half_t* p = kT + (size_t)row * 8192;
  float s = 0.f;
  for (int i = threadIdx.x; i < 8192; i += 256) s += (float)p[i];
  red[threadIdx.x] = s;
  __syncthreads();
  for (int off = 128; off > 0; off >>= 1) {
    if (threadIdx.x < off) red[threadIdx.x] += red[threadIdx.x + off];
    __syncthreads();
  }
  if (threadIdx.x == 0) ksum[row] = red[0];
}

// ---------------- Kernel 3: kv[bh] = kT @ v (64x8192 @ 8192x64) -------------
__global__ __launch_bounds__(256) void la_kv_kernel(
    const half_t* __restrict__ kT, const half_t* __restrict__ v,
    float* __restrict__ kv) {
  __shared__ half_t As[64 * 136];
  __shared__ half_t Bs[64 * 136];
  const int tid = threadIdx.x, lane = tid & 31, wid = tid >> 5;
  const int bh = blockIdx.x;
  const int mt = (wid >> 1) * 16;              // d-tile per wave
  const int nt0 = (wid & 1) * 2;               // 2 e-tiles per wave
  v8f acc[2];
  acc[0] = vzero8(); acc[1] = vzero8();
  const half_t* kbase = kT + (size_t)bh * 64 * 8192;
  const half_t* vbase = v + (size_t)bh * 8192 * 64;

  for (int n0 = 0; n0 < 8192; n0 += 128) {
    __syncthreads();
#pragma unroll
    for (int p = 0; p < 4; ++p) {     // kT tile 64x128: async pure copy -> LDS
      int c = p * 256 + tid;
      int dr = c >> 4, c8 = c & 15;
      async_copy16(&As[dr * 136 + c8 * 8], kbase + (size_t)dr * 8192 + n0 + c8 * 8);
    }
#pragma unroll
    for (int p = 0; p < 4; ++p) {     // v tile 128x64 -> N-major (transpose, VGPR path)
      int c = p * 256 + tid;
      int nr = c >> 3, e8 = c & 7;
      v8h vv = *(const v8h*)(vbase + (size_t)(n0 + nr) * 64 + e8 * 8);
#pragma unroll
      for (int j = 0; j < 8; ++j) Bs[(e8 * 8 + j) * 136 + nr] = vv[j];
    }
    wait_async();
    __syncthreads();
#pragma unroll
    for (int ks = 0; ks < 128; ks += 32) {
      v16h a = frag_a(As, 136, mt, ks, lane);
#pragma unroll
      for (int ni = 0; ni < 2; ++ni) {
        v16h b = frag_b(Bs, 136, (nt0 + ni) * 16, ks, lane);
        acc[ni] = wmma16(a, b, acc[ni]);
      }
    }
  }
#pragma unroll
  for (int ni = 0; ni < 2; ++ni)
#pragma unroll
    for (int r = 0; r < 8; ++r) {
      int d = mt + r + ((lane & 16) ? 8 : 0);
      int e = (nt0 + ni) * 16 + (lane & 15);
      kv[(size_t)bh * 4096 + d * 64 + e] = acc[ni][r];
    }
}

// ---------------- Kernel 4: out = (q @ kv) * z, z = 1/(q.ksum + eps) --------
__global__ __launch_bounds__(256) void la_attn_kernel(
    const half_t* __restrict__ q, const float* __restrict__ kv,
    const float* __restrict__ ksum, half_t* __restrict__ attn) {
  __shared__ half_t As[128 * 72];
  __shared__ half_t Bs[64 * 72];
  __shared__ float zs[128];
  __shared__ float ks_s[64];
  const int tid = threadIdx.x, lane = tid & 31, wid = tid >> 5;
  const int bh = blockIdx.x >> 6;
  const int n0 = (blockIdx.x & 63) * 128;
  const half_t* qbase = q + ((size_t)bh * 8192 + n0) * 64;

#pragma unroll
  for (int p = 0; p < 4; ++p) {                // q tile 128x64: async pure copy
    int c = p * 256 + tid;
    int nr = c >> 3, d8 = c & 7;
    async_copy16(&As[nr * 72 + d8 * 8], qbase + (size_t)nr * 64 + d8 * 8);
  }
#pragma unroll
  for (int p = 0; p < 16; ++p) {               // kv 64x64 f32 -> N-major f16
    int idx = p * 256 + tid;
    int d = idx >> 6, e = idx & 63;
    Bs[e * 72 + d] = (half_t)kv[(size_t)bh * 4096 + idx];
  }
  if (tid < 64) ks_s[tid] = ksum[bh * 64 + tid];
  wait_async();
  __syncthreads();
  if (tid < 128) {                             // per-row normalizer
    float s = 0.f;
#pragma unroll
    for (int d = 0; d < 64; ++d) s += (float)As[tid * 72 + d] * ks_s[d];
    zs[tid] = 1.f / (s + 1e-8f);
  }
  __syncthreads();

  v8f acc[4];
#pragma unroll
  for (int ni = 0; ni < 4; ++ni) acc[ni] = vzero8();
#pragma unroll
  for (int ksx = 0; ksx < 64; ksx += 32) {
    v16h a = frag_a(As, 72, wid * 16, ksx, lane);
#pragma unroll
    for (int ni = 0; ni < 4; ++ni) {
      v16h b = frag_b(Bs, 72, ni * 16, ksx, lane);
      acc[ni] = wmma16(a, b, acc[ni]);
    }
  }
  const int b_ = bh >> 4, h_ = bh & 15;
#pragma unroll
  for (int ni = 0; ni < 4; ++ni)
#pragma unroll
    for (int r = 0; r < 8; ++r) {
      int nl = wid * 16 + r + ((lane & 16) ? 8 : 0);
      int e = ni * 16 + (lane & 15);
      float o = acc[ni][r] * zs[nl];
      attn[((size_t)b_ * 8192 + n0 + nl) * 1024 + h_ * 64 + e] = (half_t)o;
    }
}

// ---------------- Kernel 5: proj GEMM: attn[32768,1024] @ w[1024,1024] ------
__global__ __launch_bounds__(256) void la_proj_kernel(
    const half_t* __restrict__ a, const float* __restrict__ w,
    const float* __restrict__ bias, float* __restrict__ out) {
  __shared__ half_t As[128 * 72];
  __shared__ half_t Bs[128 * 72];
  const int tid = threadIdx.x, lane = tid & 31, wid = tid >> 5;
  const int wm = wid >> 1, wn = wid & 1;
  const int m0 = blockIdx.y * 128, n0 = blockIdx.x * 128;

  v8f acc[2][4];
#pragma unroll
  for (int mi = 0; mi < 2; ++mi)
#pragma unroll
    for (int ni = 0; ni < 4; ++ni) acc[mi][ni] = vzero8();

  for (int kt = 0; kt < 1024; kt += 64) {
    __syncthreads();
    if (kt + 64 < 1024) {
      __builtin_prefetch(w + (size_t)(kt + 64 + (tid >> 5)) * 1024 + n0 + (tid & 31) * 4, 0, 1);
    }
#pragma unroll
    for (int p = 0; p < 4; ++p) {     // A tile 128x64 f16: async pure copy
      int c = p * 256 + tid;
      int row = c >> 3, k8 = c & 7;
      async_copy16(&As[row * 72 + k8 * 8], a + (size_t)(m0 + row) * 1024 + kt + k8 * 8);
    }
#pragma unroll
    for (int p = 0; p < 8; ++p) {     // B tile 64x128 f32 -> N-major f16
      int f = p * 256 + tid;
      int kr = f >> 5, c4 = f & 31;
      float4 wv = *(const float4*)(w + (size_t)(kt + kr) * 1024 + n0 + c4 * 4);
      Bs[(c4 * 4 + 0) * 72 + kr] = (half_t)wv.x;
      Bs[(c4 * 4 + 1) * 72 + kr] = (half_t)wv.y;
      Bs[(c4 * 4 + 2) * 72 + kr] = (half_t)wv.z;
      Bs[(c4 * 4 + 3) * 72 + kr] = (half_t)wv.w;
    }
    wait_async();
    __syncthreads();
#pragma unroll
    for (int ks = 0; ks < 64; ks += 32) {
      v16h af[2], bf[4];
#pragma unroll
      for (int mi = 0; mi < 2; ++mi) af[mi] = frag_a(As, 72, wm * 32 + mi * 16, ks, lane);
#pragma unroll
      for (int ni = 0; ni < 4; ++ni) bf[ni] = frag_b(Bs, 72, wn * 64 + ni * 16, ks, lane);
#pragma unroll
      for (int mi = 0; mi < 2; ++mi)
#pragma unroll
        for (int ni = 0; ni < 4; ++ni) acc[mi][ni] = wmma16(af[mi], bf[ni], acc[mi][ni]);
    }
  }
#pragma unroll
  for (int mi = 0; mi < 2; ++mi)
#pragma unroll
    for (int ni = 0; ni < 4; ++ni)
#pragma unroll
      for (int r = 0; r < 8; ++r) {
        int row = m0 + wm * 32 + mi * 16 + r + ((lane & 16) ? 8 : 0);
        int col = n0 + wn * 64 + ni * 16 + (lane & 15);
        out[(size_t)row * 1024 + col] = acc[mi][ni][r] + bias[col];
      }
}

extern "C" void kernel_launch(void* const* d_in, const int* in_sizes, int n_in,
                              void* d_out, int out_size, void* d_ws, size_t ws_size,
                              hipStream_t stream) {
  (void)in_sizes; (void)n_in; (void)out_size; (void)ws_size;
  const float* x      = (const float*)d_in[0];
  const float* w_qkv  = (const float*)d_in[1];
  const float* b_qkv  = (const float*)d_in[2];
  const float* w_proj = (const float*)d_in[3];
  const float* b_proj = (const float*)d_in[4];
  float* out = (float*)d_out;

  char* ws = (char*)d_ws;
  const size_t SZ = 67108864ull;                 // 64 MiB per f16 [B,H,N,D] tensor
  half_t* q    = (half_t*)(ws);
  half_t* kT   = (half_t*)(ws + SZ);
  half_t* v    = (half_t*)(ws + 2 * SZ);
  half_t* attn = v;                              // v fully consumed before attn written
  float*  kv   = (float*)(ws + 3 * SZ);          // 64*64*64 f32 = 1 MiB
  float*  ksum = (float*)(ws + 3 * SZ + (1 << 20));
  // total workspace use: 3*64 MiB + ~1.1 MiB = ~193.1 MiB

  la_qkv_kernel<<<dim3(24, 256), 256, 0, stream>>>(x, w_qkv, b_qkv, q, kT, v);
  la_ksum_kernel<<<4096, 256, 0, stream>>>(kT, ksum);
  la_kv_kernel<<<64, 256, 0, stream>>>(kT, v, kv);
  la_attn_kernel<<<4096, 256, 0, stream>>>(q, kv, ksum, attn);
  la_proj_kernel<<<dim3(8, 256), 256, 0, stream>>>(attn, w_proj, b_proj, out);
}
